// GATv2SelfAttention_2697239462694
// MI455X (gfx1250) — matrix operand order repaired and checked by reading, not compiled, not run on hardware
//
#include <hip/hip_runtime.h>
#include <hip/hip_bf16.h>

// GATv2 self-attention for MI455X (gfx1250, wave32, WMMA).
// 8 GEMMs [2048x4096]x[4096x2048] = 275 GFLOP, AI ~950 FLOP/B -> compute
// bound -> bf16 WMMA w/ f32 accum. All f32->bf16 conversion hoisted into
// memory-bound pre-passes; GEMM K-loop is async-to-LDS double-buffered,
// rolled as a ping-pong pair loop to keep the I$ footprint small.

#define ALPHA 0.3f

constexpr int Bq = 256, Sq = 8, Fq = 2048, Hq = 8;
constexpr int Mdim = Bq * Sq;   // 2048 rows (b*8+s)
constexpr int Kdim = 2 * Fq;    // 4096
constexpr int Ndim = Fq;        // 2048
constexpr int BM = 128, BN = 128, BK = 32;
constexpr int KSTEPS = Kdim / BK;   // 128

typedef __attribute__((ext_vector_type(16))) __bf16       bf16x16;
typedef __attribute__((ext_vector_type(8)))  float        floatx8;
typedef __attribute__((ext_vector_type(4)))  unsigned int u32x4;
typedef __attribute__((ext_vector_type(2)))  unsigned int u32x2;

__device__ __forceinline__ unsigned int f32_to_bf16(float x) {
  union { float f; unsigned int u; } v; v.f = x;
  unsigned int u = v.u;
  u += 0x7FFFu + ((u >> 16) & 1u);   // round-to-nearest-even
  return u >> 16;
}

// ---------------------------------------------------------------------------
// Pre-pass 1: A = bf16(h_cat) as [M][K] row-major.  16 MB, memory-bound.
// ---------------------------------------------------------------------------
__global__ __launch_bounds__(256)
void conv_a(const float* __restrict__ inputs, unsigned short* __restrict__ A) {
  size_t i = (size_t)blockIdx.x * 256 + threadIdx.x;   // one float4 each
  int k4 = (int)(i % (Kdim / 4)) * 4;
  int m  = (int)(i / (Kdim / 4));
  int b = m >> 3, s = m & 7;
  const float* src = (k4 < Fq)
      ? inputs + (size_t)b * Sq * Fq + k4                 // node-0 broadcast
      : inputs + ((size_t)b * Sq + s) * Fq + (k4 - Fq);
  float4 v = *(const float4*)src;
  u32x2 p;
  p.x = f32_to_bf16(v.x) | (f32_to_bf16(v.y) << 16);
  p.y = f32_to_bf16(v.z) | (f32_to_bf16(v.w) << 16);
  *(u32x2*)&A[(size_t)m * Kdim + k4] = p;
}

// ---------------------------------------------------------------------------
// Pre-pass 2: Wt[h][n][k] = bf16(W[h][k][n])  (transpose via LDS tile).
// ---------------------------------------------------------------------------
__global__ __launch_bounds__(256)
void conv_w(const float* __restrict__ W, unsigned short* __restrict__ Wt) {
  __shared__ unsigned short t[64][72];       // +8 pad vs bank conflicts
  const int k0 = blockIdx.x * 64, n0 = blockIdx.y * 64, h = blockIdx.z;
  const float* Wb = W + (size_t)h * Kdim * Ndim;
  const int tid = threadIdx.x;
#pragma unroll
  for (int tt = 0; tt < 4; ++tt) {
    int c = tid + tt * 256;                  // 0..1023 float4 chunks
    int row  = c >> 4;                       // k local
    int col4 = (c & 15) << 2;                // n local
    float4 v = *(const float4*)(Wb + (size_t)(k0 + row) * Ndim + n0 + col4);
    u32x2 p;
    p.x = f32_to_bf16(v.x) | (f32_to_bf16(v.y) << 16);
    p.y = f32_to_bf16(v.z) | (f32_to_bf16(v.w) << 16);
    *(__shared__ u32x2*)&t[row][col4] = p;
  }
  __syncthreads();
#pragma unroll
  for (int tt = 0; tt < 4; ++tt) {
    int c = tid + tt * 256;
    int row  = c >> 4;                       // n local
    int col4 = (c & 15) << 2;                // k local
    unsigned int a0 = t[col4 + 0][row], a1 = t[col4 + 1][row];
    unsigned int a2 = t[col4 + 2][row], a3 = t[col4 + 3][row];
    u32x2 p; p.x = a0 | (a1 << 16); p.y = a2 | (a3 << 16);
    *(u32x2*)(Wt + (size_t)h * Ndim * Kdim +
              (size_t)(n0 + row) * Kdim + k0 + col4) = p;
  }
}

// ---------------------------------------------------------------------------
// Kernel 1: Wh[h][m][n] = leaky_relu( A @ Wt[h]^T ), bf16 WMMA core.
// ---------------------------------------------------------------------------
__global__ __launch_bounds__(256)
void gatv2_gemm_wh(const unsigned short* __restrict__ A,    // [M][K] bf16
                   const unsigned short* __restrict__ Wt,   // [H][N][K] bf16
                   float* __restrict__ Wh) {                // [H][M][N] f32
  __shared__ unsigned short lds[2][(BM + BN) * BK];   // per buf: A | B, 16KB

  const int n0 = blockIdx.x * BN, m0 = blockIdx.y * BM, h = blockIdx.z;
  const int tid = threadIdx.x, lane = tid & 31, wid = tid >> 5;
  const int wm = (wid >> 2) * 64;            // wave M base: 0/64
  const int wn = (wid & 3) * 32;             // wave N base: 0/32/64/96
  const int lm = lane & 15, half = lane >> 4;
  const unsigned short* Wth = Wt + (size_t)h * Ndim * Kdim;

  // Per-thread staging: 2x b128 chunks for A tile, 2x for B tile.
  unsigned voffA[2], voffB[2], ldsAo[2][2], ldsBo[2][2];
#pragma unroll
  for (int t = 0; t < 2; ++t) {
    int c = tid + t * 256, row = c >> 2, q = c & 3;
    voffA[t] = (unsigned)((m0 + row) * Kdim * 2 + q * 16);
    voffB[t] = (unsigned)((n0 + row) * Kdim * 2 + q * 16);
#pragma unroll
    for (int bu = 0; bu < 2; ++bu) {
      ldsAo[bu][t] = (unsigned)(size_t)&lds[bu][0]       + (unsigned)c * 16;
      ldsBo[bu][t] = (unsigned)(size_t)&lds[bu][BM * BK] + (unsigned)c * 16;
    }
  }

  auto issue_tile = [&](int it, int bu) {
    unsigned ko = (unsigned)(it * BK * 2);   // byte offset along K
#pragma unroll
    for (int t = 0; t < 2; ++t) {
      asm volatile("global_load_async_to_lds_b128 %0, %1, %2 offset:0"
                   :: "v"(ldsAo[bu][t]), "v"(voffA[t] + ko), "s"(A)
                   : "memory");
      asm volatile("global_load_async_to_lds_b128 %0, %1, %2 offset:0"
                   :: "v"(ldsBo[bu][t]), "v"(voffB[t] + ko), "s"(Wth)
                   : "memory");
    }
  };

  floatx8 zero = {};
  floatx8 acc[4][2];
#pragma unroll
  for (int i = 0; i < 4; ++i)
#pragma unroll
    for (int j = 0; j < 2; ++j) acc[i][j] = zero;

  const int hiA = half ? 8 : 0;    // ISA 7.12.2: lanes16-31 K 8..15/24..31
  const int hiB = half ? 8 : 0;    //              lanes16-31 K 8..15/24..31

  auto compute_tile = [&](int bu) {
    union Frag { bf16x16 v; u32x4 q[2]; };
    Frag a[4], bfr[2];
    const unsigned short* Abuf = &lds[bu][0];
    const unsigned short* Bbuf = &lds[bu][BM * BK];
#pragma unroll
    for (int i = 0; i < 4; ++i) {
      const unsigned short* ap = Abuf + (wm + i * 16 + lm) * BK;
      a[i].q[0] = *(const __shared__ u32x4*)(ap + hiA);
      a[i].q[1] = *(const __shared__ u32x4*)(ap + hiA + 16);
    }
#pragma unroll
    for (int j = 0; j < 2; ++j) {
      const unsigned short* bp = Bbuf + (wn + j * 16 + lm) * BK + hiB;
      bfr[j].q[0] = *(const __shared__ u32x4*)(bp);
      bfr[j].q[1] = *(const __shared__ u32x4*)(bp + 16);
    }
#pragma unroll
    for (int i = 0; i < 4; ++i)
#pragma unroll
      for (int j = 0; j < 2; ++j)
        acc[i][j] = __builtin_amdgcn_wmma_f32_16x16x32_bf16(
            false, a[i].v, false, bfr[j].v, (short)0, acc[i][j], false, false);
  };

  // Ping-pong pair loop, kept rolled: compact loop body (~2x12 ds_load,
  // 2x8 wmma, 2x4 async issues) instead of a 40KB fully-unrolled stream.
  issue_tile(0, 0);
#pragma clang loop unroll(disable)
  for (int it2 = 0; it2 < KSTEPS / 2 - 1; ++it2) {
    const int it = 2 * it2;
    issue_tile(it + 1, 1);
    asm volatile("s_wait_asynccnt 0x4" ::: "memory");
    __syncthreads();
    compute_tile(0);
    __syncthreads();
    issue_tile(it + 2, 0);
    asm volatile("s_wait_asynccnt 0x4" ::: "memory");
    __syncthreads();
    compute_tile(1);
    __syncthreads();
  }
  // last pair (tiles KSTEPS-2 in buf0, KSTEPS-1 in buf1)
  issue_tile(KSTEPS - 1, 1);
  asm volatile("s_wait_asynccnt 0x4" ::: "memory");
  __syncthreads();
  compute_tile(0);
  __syncthreads();
  asm volatile("s_wait_asynccnt 0x0" ::: "memory");
  __syncthreads();
  compute_tile(1);

  // epilogue: LeakyReLU + store (C layout: VGPR r -> M=r / M=8+r)
  const int rbase = half * 8;
#pragma unroll
  for (int i = 0; i < 4; ++i)
#pragma unroll
    for (int j = 0; j < 2; ++j) {
      int col = n0 + wn + j * 16 + lm;
#pragma unroll
      for (int r = 0; r < 8; ++r) {
        float x = acc[i][j][r];
        x = x > 0.0f ? x : ALPHA * x;
        int row = m0 + wm + i * 16 + rbase + r;
        Wh[((size_t)h * Mdim + row) * Ndim + col] = x;
      }
    }
}

// ---------------------------------------------------------------------------
// Kernel 2: e[b,h,s] = Wh[h, b*8+s, :] . att_w[h]; softmax over s.
// ---------------------------------------------------------------------------
__global__ __launch_bounds__(256)
void gatv2_attn(const float* __restrict__ Wh,
                const float* __restrict__ att_w,
                float* __restrict__ attn) {           // [B,H,S]
  const int bh = blockIdx.x;
  const int b = bh / Hq, h = bh % Hq;
  __shared__ float red[256];
  __shared__ float esh[Sq];
  const float* aw = att_w + h * Fq;
  for (int s = 0; s < Sq; ++s) {
    const float* wr = Wh + ((size_t)h * Mdim + b * Sq + s) * Ndim;
    float p = 0.0f;
    for (int f = threadIdx.x; f < Fq; f += 256) p += wr[f] * aw[f];
    red[threadIdx.x] = p;
    __syncthreads();
    for (int off = 128; off > 0; off >>= 1) {
      if (threadIdx.x < off) red[threadIdx.x] += red[threadIdx.x + off];
      __syncthreads();
    }
    if (threadIdx.x == 0) esh[s] = red[0];
    __syncthreads();
  }
  if (threadIdx.x == 0) {
    float mx = esh[0];
    for (int s = 1; s < Sq; ++s) mx = fmaxf(mx, esh[s]);
    float ex[Sq], den = 0.0f;
    for (int s = 0; s < Sq; ++s) { ex[s] = __expf(esh[s] - mx); den += ex[s]; }
    for (int s = 0; s < Sq; ++s) attn[(size_t)bh * Sq + s] = ex[s] / den;
  }
}

// ---------------------------------------------------------------------------
// Kernel 3: out = inputs + (sum_h attn*Wh)/H + (sum_h bias)/H.
// ---------------------------------------------------------------------------
__global__ __launch_bounds__(256)
void gatv2_out(const float* __restrict__ inputs,
               const float* __restrict__ bias,
               const float* __restrict__ Wh,
               const float* __restrict__ attn,
               float* __restrict__ out) {
  size_t idx = (size_t)blockIdx.x * 256 + threadIdx.x;  // one float4 each
  int f4 = (int)(idx % (Fq / 4)) * 4;
  int bs = (int)(idx / (Fq / 4));
  int b = bs >> 3, s = bs & 7;
  float4 in = *(const float4*)(inputs + (size_t)bs * Fq + f4);
  float ax = 0.f, ay = 0.f, az = 0.f, aw = 0.f;
#pragma unroll
  for (int h = 0; h < Hq; ++h) {
    float a = attn[((size_t)b * Hq + h) * Sq + s];
    float4 w  = *(const float4*)(Wh + ((size_t)h * Mdim + bs) * Ndim + f4);
    float4 bi = *(const float4*)(bias + (size_t)h * Fq + f4);
    ax += a * w.x + bi.x;  ay += a * w.y + bi.y;
    az += a * w.z + bi.z;  aw += a * w.w + bi.w;
  }
  const float inv = 1.0f / Hq;
  float4 o;
  o.x = in.x + ax * inv;  o.y = in.y + ay * inv;
  o.z = in.z + az * inv;  o.w = in.w + aw * inv;
  *(float4*)(out + (size_t)bs * Fq + f4) = o;
}

// ---------------------------------------------------------------------------
extern "C" void kernel_launch(void* const* d_in, const int* in_sizes, int n_in,
                              void* d_out, int out_size, void* d_ws, size_t ws_size,
                              hipStream_t stream) {
  (void)in_sizes; (void)n_in; (void)out_size; (void)ws_size;
  const float* inputs = (const float*)d_in[0];   // [B,S,F]
  const float* W      = (const float*)d_in[1];   // [H,2F,F]
  const float* att_w  = (const float*)d_in[2];   // [H,F]
  const float* bias   = (const float*)d_in[3];   // [H,F]
  float* out = (float*)d_out;

  // Workspace: Wh f32 128MB | attn 64KB | A bf16 16MB | Wt bf16 128MB
  char* p = (char*)d_ws;
  float* Wh = (float*)p;                   p += (size_t)Hq * Mdim * Ndim * 4;
  float* attn = (float*)p;                 p += (size_t)Bq * Hq * Sq * 4;
  unsigned short* Abf = (unsigned short*)p; p += (size_t)Mdim * Kdim * 2;
  unsigned short* Wtb = (unsigned short*)p;

  conv_a<<<dim3((Mdim * Kdim / 4) / 256), 256, 0, stream>>>(inputs, Abf);
  conv_w<<<dim3(Kdim / 64, Ndim / 64, Hq), 256, 0, stream>>>(W, Wtb);

  gatv2_gemm_wh<<<dim3(Ndim / BN, Mdim / BM, Hq), 256, 0, stream>>>(Abf, Wtb, Wh);

  gatv2_attn<<<dim3(Bq * Hq), 256, 0, stream>>>(Wh, att_w, attn);

  gatv2_out<<<dim3((unsigned)((size_t)Bq * Sq * Fq / 4 / 256)), 256, 0, stream>>>(
      inputs, bias, Wh, attn, out);
}
